// SpatialEmbedding_81123342286998
// MI455X (gfx1250) — matrix operand order repaired and checked by reading, not compiled, not run on hardware
//
#include <hip/hip_runtime.h>
#include <hip/hip_bf16.h>
#include <cstdint>

#define NN 50000
#define NE 800000
#define DM 128
#define NH 4
#define FH 32
#define NL 2

typedef __attribute__((ext_vector_type(16))) __bf16 v16bf;
typedef __attribute__((ext_vector_type(8)))  __bf16 v8bf;
typedef __attribute__((ext_vector_type(8)))  float  v8f;

static __device__ __forceinline__ __bf16 f2bf(float f) {
    union { float f; uint32_t u; } v; v.f = f;
    uint32_t r = v.u + 0x7FFFu + ((v.u >> 16) & 1u);   // round-to-nearest-even
    unsigned short h = (unsigned short)(r >> 16);
    return __builtin_bit_cast(__bf16, h);
}

// ---------------- convert fp32 -> bf16 (x matrix) ----------------
__global__ void cvt_bf16_kernel(const float* __restrict__ in, __bf16* __restrict__ out, int n) {
    int i = blockIdx.x * blockDim.x + threadIdx.x;
    if (i < n) out[i] = f2bf(in[i]);
}

// ---------------- fused convert + pack W into B-fragment order ----------------
// Packed layout: wbp[((ct*4 + kt)*32 + lane)*16 + idx],
// holding W[kt*32 + kbB + (idx&7) + (idx>=8 ? 16 : 0)][ct*16 + (lane&15)], kbB = (lane<16)?0:8.
// Each lane's 16-bf16 fragment is then one contiguous 32-byte vector.
__global__ void pack_w_kernel(const float* __restrict__ W, __bf16* __restrict__ wbp) {
    int p = blockIdx.x * blockDim.x + threadIdx.x;
    if (p >= DM * DM) return;
    int idx  = p & 15;
    int lane = (p >> 4) & 31;
    int kt   = (p >> 9) & 3;
    int ct   = p >> 11;
    int j    = idx >> 1;
    int half = idx & 1;
    int kbB  = (lane < 16) ? 0 : 8;
    int kk   = kt * 32 + kbB + (idx & 7) + ((idx >> 3) << 4);
    int n    = ct * 16 + (lane & 15);
    wbp[p] = f2bf(W[kk * DM + n]);
}

// ---------------- WMMA GEMM: h[NN,128] = xb[NN,128] @ W[128,128] ----------------
// grid.x = NN/16 row tiles, block = 128 threads (4 waves).
// Wave w computes column tiles 2w, 2w+1 (16 cols each) -> 128 cols total.
__global__ void gemm_bf16_wmma_kernel(const __bf16* __restrict__ xb,
                                      const __bf16* __restrict__ wbp,
                                      float* __restrict__ h) {
    const int wave = threadIdx.x >> 5;
    const int lane = threadIdx.x & 31;
    const int row0 = blockIdx.x * 16;

    const int mloc = lane & 15;
    const int kbA  = (lane < 16) ? 0 : 8;   // ISA 16-bit A 16x32 fragment K base
    const int row  = row0 + mloc;

    const v16bf* wp = (const v16bf*)wbp;    // 1024 packed fragments
    v8f c0 = {}; v8f c1 = {};

    #pragma unroll
    for (int kt = 0; kt < 4; ++kt) {
        const int k0 = kt * 32;
        // A fragment: elements 0..7 = K[k0+kbA .. +7], 8..15 = K[k0+kbA+16 .. +23]
        union { v16bf v; v8bf hx[2]; } au;
        const v8bf* xr = (const v8bf*)(xb + row * DM + k0 + kbA);
        au.hx[0] = xr[0];
        au.hx[1] = xr[2];
        // B fragments: one 32-byte vector load each
        v16bf b0 = wp[((2 * wave)     * 4 + kt) * 32 + lane];
        v16bf b1 = wp[((2 * wave + 1) * 4 + kt) * 32 + lane];

        c0 = __builtin_amdgcn_wmma_f32_16x16x32_bf16(false, au.v, false, b0, (short)0, c0, false, false);
        c1 = __builtin_amdgcn_wmma_f32_16x16x32_bf16(false, au.v, false, b1, (short)0, c1, false, false);
    }

    // D store: VGPR i -> M = (lane<16)? i : 8+i ; N = lane&15
    const int mbase = (lane < 16) ? 0 : 8;
    #pragma unroll
    for (int i = 0; i < 8; ++i) {
        int r = row0 + mbase + i;
        h[r * DM + wave * 32 + mloc]      = c0[i];
        h[r * DM + wave * 32 + 16 + mloc] = c1[i];
    }
}

// ---------------- per-node attention dots: alpha_s/d[n][h] ----------------
__global__ void alpha_kernel(const float* __restrict__ h,
                             const float* __restrict__ a_src_l,
                             const float* __restrict__ a_dst_l,
                             float* __restrict__ as, float* __restrict__ ad) {
    int idx = blockIdx.x * blockDim.x + threadIdx.x;
    if (idx >= NN * NH) return;
    int n = idx >> 2, hh = idx & 3;
    const float* hp = h + n * DM + hh * FH;
    const float* sp = a_src_l + hh * FH;
    const float* dp = a_dst_l + hh * FH;
    float s = 0.f, d = 0.f;
    #pragma unroll
    for (int f = 0; f < FH; ++f) { s += hp[f] * sp[f]; d += hp[f] * dp[f]; }
    as[idx] = s; ad[idx] = d;
}

// ---------------- init accumulators ----------------
__global__ void init_kernel(float* __restrict__ acc, float* __restrict__ emax,
                            float* __restrict__ denom) {
    int i = blockIdx.x * blockDim.x + threadIdx.x;
    if (i < NN * DM) acc[i] = 0.f;
    if (i < NN * NH) { emax[i] = -3.0e38f; denom[i] = 0.f; }
}

static __device__ __forceinline__ void atomicMaxF(float* addr, float val) {
    if (val >= 0.f)
        atomicMax((int*)addr, __float_as_int(val));
    else
        atomicMin((unsigned int*)addr, __float_as_uint(val));
}

// ---------------- edge raw scores + segment max ----------------
__global__ void edge_score_kernel(const int* __restrict__ src, const int* __restrict__ dst,
                                  const float* __restrict__ as, const float* __restrict__ ad,
                                  float* __restrict__ ebuf, float* __restrict__ emax) {
    int idx = blockIdx.x * blockDim.x + threadIdx.x;
    if (idx >= NE * NH) return;
    int e = idx >> 2, hh = idx & 3;
    int s = src[e], d = dst[e];
    float v = as[s * NH + hh] + ad[d * NH + hh];
    v = (v > 0.f) ? v : 0.2f * v;                 // LeakyReLU(0.2)
    ebuf[idx] = v;
    atomicMaxF(&emax[d * NH + hh], v);
}

// ---------------- edge exp + segment sum ----------------
__global__ void edge_exp_kernel(const int* __restrict__ dst,
                                const float* __restrict__ emax,
                                float* __restrict__ ebuf, float* __restrict__ denom) {
    int idx = blockIdx.x * blockDim.x + threadIdx.x;
    if (idx >= NE * NH) return;
    int e = idx >> 2, hh = idx & 3;
    int d = dst[e];
    float ex = __expf(ebuf[idx] - emax[d * NH + hh]);
    ebuf[idx] = ex;
    atomicAdd(&denom[d * NH + hh], ex);
}

// ---------------- weighted scatter: one wave per edge ----------------
__global__ void edge_scatter_kernel(const int* __restrict__ src, const int* __restrict__ dst,
                                    const float* __restrict__ h, const float* __restrict__ ebuf,
                                    const float* __restrict__ denom, float* __restrict__ acc) {
    int wave = threadIdx.x >> 5;
    int lane = threadIdx.x & 31;
    int e = blockIdx.x * 8 + wave;
    if (e >= NE) return;
    int s = src[e], d = dst[e];
    #pragma unroll
    for (int hh = 0; hh < NH; ++hh) {
        float alpha = ebuf[e * NH + hh] / (denom[d * NH + hh] + 1e-16f);
        int col = hh * FH + lane;
        atomicAdd(&acc[d * DM + col], h[s * DM + col] * alpha);
    }
}

// ---------------- finalize: skip + bias + optional ELU ----------------
__global__ void finalize_kernel(const float* __restrict__ acc, const float* __restrict__ xin,
                                const float* __restrict__ b_l, float* __restrict__ out,
                                int apply_act) {
    int i = blockIdx.x * blockDim.x + threadIdx.x;
    if (i >= NN * DM) return;
    float v = acc[i] + xin[i] + b_l[i & (DM - 1)];
    if (apply_act) v = (v > 0.f) ? v : (__expf(v) - 1.f);  // ELU
    out[i] = v;
}

static inline size_t alignup(size_t x) { return (x + 255) & ~size_t(255); }

extern "C" void kernel_launch(void* const* d_in, const int* in_sizes, int n_in,
                              void* d_out, int out_size, void* d_ws, size_t ws_size,
                              hipStream_t stream) {
    const float* x0    = (const float*)d_in[0];
    const int*   eidx  = (const int*)d_in[1];      // [2, NE] int32
    const float* W     = (const float*)d_in[2];    // [NL,128,128]
    const float* a_src = (const float*)d_in[3];    // [NL,4,32]
    const float* a_dst = (const float*)d_in[4];
    const float* bias  = (const float*)d_in[5];    // [NL,128]
    const int* src = eidx;
    const int* dst = eidx + NE;

    char* ws = (char*)d_ws;
    size_t off = 0;
    __bf16* xb    = (__bf16*)(ws + off); off += alignup((size_t)NN * DM * 2);
    __bf16* wbp   = (__bf16*)(ws + off); off += alignup((size_t)DM * DM * 2);
    float*  h     = (float*)(ws + off);  off += alignup((size_t)NN * DM * 4);
    float*  as    = (float*)(ws + off);  off += alignup((size_t)NN * NH * 4);
    float*  ad    = (float*)(ws + off);  off += alignup((size_t)NN * NH * 4);
    float*  emax  = (float*)(ws + off);  off += alignup((size_t)NN * NH * 4);
    float*  denom = (float*)(ws + off);  off += alignup((size_t)NN * NH * 4);
    float*  ebuf  = (float*)(ws + off);  off += alignup((size_t)NE * NH * 4);
    float*  acc   = (float*)(ws + off);  off += alignup((size_t)NN * DM * 4);
    float*  xcur  = (float*)(ws + off);  off += alignup((size_t)NN * DM * 4);

    for (int l = 0; l < NL; ++l) {
        const float* x_in = (l == 0) ? x0 : xcur;
        float* x_out = (l == NL - 1) ? (float*)d_out : xcur;

        cvt_bf16_kernel<<<(NN * DM + 255) / 256, 256, 0, stream>>>(x_in, xb, NN * DM);
        pack_w_kernel<<<(DM * DM + 255) / 256, 256, 0, stream>>>(W + (size_t)l * DM * DM, wbp);
        gemm_bf16_wmma_kernel<<<NN / 16, 128, 0, stream>>>(xb, wbp, h);
        alpha_kernel<<<(NN * NH + 255) / 256, 256, 0, stream>>>(
            h, a_src + (size_t)l * NH * FH, a_dst + (size_t)l * NH * FH, as, ad);
        init_kernel<<<(NN * DM + 255) / 256, 256, 0, stream>>>(acc, emax, denom);
        edge_score_kernel<<<(NE * NH + 255) / 256, 256, 0, stream>>>(src, dst, as, ad, ebuf, emax);
        edge_exp_kernel<<<(NE * NH + 255) / 256, 256, 0, stream>>>(dst, emax, ebuf, denom);
        edge_scatter_kernel<<<(NE + 7) / 8, 256, 0, stream>>>(src, dst, h, ebuf, denom, acc);
        finalize_kernel<<<(NN * DM + 255) / 256, 256, 0, stream>>>(
            acc, x_in, bias + (size_t)l * DM, x_out, (l < NL - 1) ? 1 : 0);
    }
}
